// HWAttention_16896401343236
// MI455X (gfx1250) — matrix operand order repaired and checked by reading, not compiled, not run on hardware
//
#include <hip/hip_runtime.h>
#include <hip/hip_bf16.h>

typedef __attribute__((ext_vector_type(16))) _Float16 v16h;
typedef __attribute__((ext_vector_type(8)))  _Float16 v8h;
typedef __attribute__((ext_vector_type(2)))  _Float16 v2h;
typedef __attribute__((ext_vector_type(8)))  float    v8f;

namespace {
constexpr int   kB = 4, kN = 2048, kC3 = 1536, kC = 512, kH = 16, kD = 32;
// fold 1/sqrt(d) AND log2(e) into the Q pre-scale: softmax runs in exp2 domain
constexpr float kScaleLog2e = 0.1767766952966369f * 1.4426950408889634f;
constexpr int   kKChunk  = 64;                  // keys staged per iteration
constexpr int   kStrideK = 40;                  // sK row stride in halfs (32+8 pad)
constexpr int   kStrideP = 72;                  // sVT/sP row stride in halfs (64+8 pad)
constexpr int   kWaves   = 4;                   // 64 query rows per block
constexpr int   kNIter   = kN / kKChunk;        // 32
constexpr int   kSKBuf   = kKChunk * kStrideK;  // halfs per sK buffer
constexpr int   kSVBuf   = kD * kStrideP;       // halfs per sVT buffer
}

// raw VOP2 max: avoids compiler-inserted canonicalizing self-max
__device__ __forceinline__ float fmax_raw(float a, float b) {
  float d;
  asm("v_max_num_f32_e32 %0, %1, %2" : "=v"(d) : "v"(a), "v"(b));
  return d;
}

// fused shuffle+max butterfly over a 16-lane DPP row: 1 instruction per step
__device__ __forceinline__ float row_max16(float t) {
  asm("v_max_num_f32_dpp %0, %1, %1 quad_perm:[1,0,3,2] row_mask:0xf bank_mask:0xf bound_ctrl:1"
      : "=v"(t) : "v"(t));   // xor 1
  asm("v_max_num_f32_dpp %0, %1, %1 quad_perm:[2,3,0,1] row_mask:0xf bank_mask:0xf bound_ctrl:1"
      : "=v"(t) : "v"(t));   // xor 2
  asm("v_max_num_f32_dpp %0, %1, %1 row_half_mirror row_mask:0xf bank_mask:0xf bound_ctrl:1"
      : "=v"(t) : "v"(t));   // xor 4 (quads uniform)
  asm("v_max_num_f32_dpp %0, %1, %1 row_mirror row_mask:0xf bank_mask:0xf bound_ctrl:1"
      : "=v"(t) : "v"(t));   // xor 8 (octets uniform)
  return t;
}

// A fragment, contiguous k order used here (not AMD's table order): lane<16 holds row L%16 cols [0..15]; lane>=16 cols [16..31]
__device__ __forceinline__ v16h lds_a_frag(const _Float16* rowp, int hsel) {
  v8h x0 = *(const v8h*)(rowp + 16 * hsel);
  v8h x1 = *(const v8h*)(rowp + 16 * hsel + 8);
  v16h r;
#pragma unroll
  for (int i = 0; i < 8; ++i) { r[i] = x0[i]; r[i + 8] = x1[i]; }
  return r;
}

// B fragment, same contiguous k order as the A fragments: lane holds column N = L%16; 16 consecutive reduction-K halfs (lanes>=16 take K+16)
__device__ __forceinline__ v16h lds_b_frag(const _Float16* p) {
  v8h x0 = *(const v8h*)(p);
  v8h x1 = *(const v8h*)(p + 8);
  v16h r;
#pragma unroll
  for (int i = 0; i < 8; ++i) { r[i] = x0[i]; r[i + 8] = x1[i]; }
  return r;
}

struct Chunk { float4 k0, k1, k2, k3, w0, w1, w2, w3; };

__device__ __forceinline__ Chunk load_chunk(const float* gk, const float* gv) {
  Chunk c;
  c.k0 = ((const float4*)gk)[0]; c.k1 = ((const float4*)gk)[1];
  c.k2 = ((const float4*)gk)[2]; c.k3 = ((const float4*)gk)[3];
  c.w0 = ((const float4*)gv)[0]; c.w1 = ((const float4*)gv)[1];
  c.w2 = ((const float4*)gv)[2]; c.w3 = ((const float4*)gv)[3];
  return c;
}

__device__ __forceinline__ void store_chunk(const Chunk& c, _Float16* sKb, _Float16* sVTb,
                                            int skey, int sdim, int vpos) {
  v8h h0, h1;
  h0[0]=(_Float16)c.k0.x; h0[1]=(_Float16)c.k0.y; h0[2]=(_Float16)c.k0.z; h0[3]=(_Float16)c.k0.w;
  h0[4]=(_Float16)c.k1.x; h0[5]=(_Float16)c.k1.y; h0[6]=(_Float16)c.k1.z; h0[7]=(_Float16)c.k1.w;
  h1[0]=(_Float16)c.k2.x; h1[1]=(_Float16)c.k2.y; h1[2]=(_Float16)c.k2.z; h1[3]=(_Float16)c.k2.w;
  h1[4]=(_Float16)c.k3.x; h1[5]=(_Float16)c.k3.y; h1[6]=(_Float16)c.k3.z; h1[7]=(_Float16)c.k3.w;
  *(v8h*)(sKb + skey * kStrideK + sdim)     = h0;
  *(v8h*)(sKb + skey * kStrideK + sdim + 8) = h1;
  float vv[16] = {c.w0.x,c.w0.y,c.w0.z,c.w0.w, c.w1.x,c.w1.y,c.w1.z,c.w1.w,
                  c.w2.x,c.w2.y,c.w2.z,c.w2.w, c.w3.x,c.w3.y,c.w3.z,c.w3.w};
#pragma unroll
  for (int i = 0; i < 16; ++i)
    sVTb[(sdim + i) * kStrideP + vpos] = (_Float16)vv[i];
}

__global__ __launch_bounds__(128, 2)
void fa32_wmma_kernel(const float* __restrict__ x, float* __restrict__ out) {
  __shared__ _Float16 sK [2 * kSKBuf];               // double-buffered K chunk [key][dim]
  __shared__ _Float16 sVT[2 * kSVBuf];               // double-buffered V^T [dim][permuted key pos]
  __shared__ _Float16 sP [kWaves * 16 * kStrideP];   // per-wave P tile [row][permuted key pos]

  const int b    = blockIdx.z;
  const int h    = blockIdx.y;
  const int q0   = blockIdx.x * (16 * kWaves);
  const int tid  = threadIdx.x;
  const int wid  = tid >> 5;
  const int lane = tid & 31;
  const int ln16 = lane & 15;
  const int hsel = lane >> 4;

  // ---- load Q fragment (A layout), pre-scaled by log2(e)/sqrt(d) ----
  const int    qrow = q0 + wid * 16 + ln16;
  const float* qp   = x + (size_t)(b * kN + qrow) * kC3 + h * kD;
  v16h qa;
  {
    float4 f0 = *(const float4*)(qp + 16 * hsel);
    float4 f1 = *(const float4*)(qp + 16 * hsel + 4);
    float4 f2 = *(const float4*)(qp + 16 * hsel + 8);
    float4 f3 = *(const float4*)(qp + 16 * hsel + 12);
    qa[0]=(_Float16)(f0.x*kScaleLog2e);  qa[1]=(_Float16)(f0.y*kScaleLog2e);
    qa[2]=(_Float16)(f0.z*kScaleLog2e);  qa[3]=(_Float16)(f0.w*kScaleLog2e);
    qa[4]=(_Float16)(f1.x*kScaleLog2e);  qa[5]=(_Float16)(f1.y*kScaleLog2e);
    qa[6]=(_Float16)(f1.z*kScaleLog2e);  qa[7]=(_Float16)(f1.w*kScaleLog2e);
    qa[8]=(_Float16)(f2.x*kScaleLog2e);  qa[9]=(_Float16)(f2.y*kScaleLog2e);
    qa[10]=(_Float16)(f2.z*kScaleLog2e); qa[11]=(_Float16)(f2.w*kScaleLog2e);
    qa[12]=(_Float16)(f3.x*kScaleLog2e); qa[13]=(_Float16)(f3.y*kScaleLog2e);
    qa[14]=(_Float16)(f3.z*kScaleLog2e); qa[15]=(_Float16)(f3.w*kScaleLog2e);
  }

  // all-ones B fragment: P x Ones puts rowsum(P) in every column -> denominator in a WMMA
  v16h ones;
#pragma unroll
  for (int i = 0; i < 16; ++i) ones[i] = (_Float16)1.0f;

  v8f o0 = {}, o1 = {}, o2 = {};   // o2 = running denominator (per row, lane-replicated)
  float m[8];
#pragma unroll
  for (int v = 0; v < 8; ++v) m[v] = -3.0e38f;

  // staging: each thread owns one half-row (16 dims) of one key, for K and V
  const int skey = tid >> 1;
  const int sdim = (tid & 1) * 16;
  const int grp  = skey >> 4;   // 0..3
  // permuted key position matching packed P stores: 32*(k>=32) + 2*(k%16) + ((k>>4)&1)
  const int vpos = ((grp & 2) << 4) + 2 * (skey & 15) + (grp & 1);
  _Float16* myP  = sP + (wid * 16) * kStrideP;

  const float* gkBase = x + (size_t)(b * kN + skey) * kC3 + kC + h * kD + sdim;
  const float* gvBase = gkBase + kC;      // V sits kC floats after K in the fused qkv row

  // ---- prologue: stage chunk 0 into buffer 0 ----
  {
    Chunk c0 = load_chunk(gkBase, gvBase);
    store_chunk(c0, sK, sVT, skey, sdim, vpos);
  }
  __syncthreads();

  for (int it = 0; it < kNIter; ++it) {
    _Float16* sKb  = sK  + (it & 1) * kSKBuf;
    _Float16* sVTb = sVT + (it & 1) * kSVBuf;
    const bool have_next = (it + 1) < kNIter;

    // ---- issue next chunk's global loads BEFORE compute (latency hides under WMMAs) ----
    Chunk nxt;
    if (have_next) {
      const size_t off = (size_t)(it + 1) * kKChunk * kC3;
      nxt = load_chunk(gkBase + off, gvBase + off);
      if (it + 2 < kNIter) {                  // warm L2 two chunks ahead
        __builtin_prefetch(gkBase + off + (size_t)kKChunk * kC3, 0, 3);
        __builtin_prefetch(gvBase + off + (size_t)kKChunk * kC3, 0, 3);
      }
    }

    // ---- S = (Q*scale*log2e) · K^T : four 16x16 tiles covering 64 keys ----
    v16h bk0 = lds_b_frag(sKb + (ln16)      * kStrideK + 16 * hsel);
    v16h bk1 = lds_b_frag(sKb + (16 + ln16) * kStrideK + 16 * hsel);
    v16h bk2 = lds_b_frag(sKb + (32 + ln16) * kStrideK + 16 * hsel);
    v16h bk3 = lds_b_frag(sKb + (48 + ln16) * kStrideK + 16 * hsel);
    v8f z = {};
    v8f s0 = __builtin_amdgcn_wmma_f32_16x16x32_f16(false, qa, false, bk0, (short)0, z, false, false);
    v8f s1 = __builtin_amdgcn_wmma_f32_16x16x32_f16(false, qa, false, bk1, (short)0, z, false, false);
    v8f s2 = __builtin_amdgcn_wmma_f32_16x16x32_f16(false, qa, false, bk2, (short)0, z, false, false);
    v8f s3 = __builtin_amdgcn_wmma_f32_16x16x32_f16(false, qa, false, bk3, (short)0, z, false, false);

    // ---- online softmax (exp2 domain): one butterfly + one alpha per row per 64 keys ----
#pragma unroll
    for (int v = 0; v < 8; ++v) {
      float t  = fmax_raw(fmax_raw(s0[v], s1[v]), fmax_raw(s2[v], s3[v]));
      t        = row_max16(t);
      float mn = fmax_raw(m[v], t);
      float al = exp2f(m[v] - mn);
      m[v] = mn;
      float p0 = exp2f(s0[v] - mn);
      float p1 = exp2f(s1[v] - mn);
      float p2 = exp2f(s2[v] - mn);
      float p3 = exp2f(s3[v] - mn);
      o0[v] *= al;
      o1[v] *= al;
      o2[v] *= al;                                   // denominator rescales like O
      _Float16* pr = myP + (v + 8 * hsel) * kStrideP + 2 * ln16;
      v2h pA; pA[0] = (_Float16)p0; pA[1] = (_Float16)p1;
      v2h pB; pB[0] = (_Float16)p2; pB[1] = (_Float16)p3;
      *(v2h*)(pr)      = pA;                         // key pos 2c, 2c+1
      *(v2h*)(pr + 32) = pB;                         // key pos 32+2c, 32+2c+1
    }

    // ---- O += P · V over K=64 ; denominator += P · Ones ----
    v16h pa0  = lds_a_frag(myP + ln16 * kStrideP,      hsel);   // key pos 0..31
    v16h pa1  = lds_a_frag(myP + ln16 * kStrideP + 32, hsel);   // key pos 32..63
    v16h bv00 = lds_b_frag(sVTb + (ln16)      * kStrideP + 16 * hsel);        // dims 0..15,  pos 0..31
    v16h bv10 = lds_b_frag(sVTb + (16 + ln16) * kStrideP + 16 * hsel);        // dims 16..31, pos 0..31
    v16h bv01 = lds_b_frag(sVTb + (ln16)      * kStrideP + 32 + 16 * hsel);   // dims 0..15,  pos 32..63
    v16h bv11 = lds_b_frag(sVTb + (16 + ln16) * kStrideP + 32 + 16 * hsel);   // dims 16..31, pos 32..63
    o0 = __builtin_amdgcn_wmma_f32_16x16x32_f16(false, pa0, false, bv00, (short)0, o0, false, false);
    o0 = __builtin_amdgcn_wmma_f32_16x16x32_f16(false, pa1, false, bv01, (short)0, o0, false, false);
    o1 = __builtin_amdgcn_wmma_f32_16x16x32_f16(false, pa0, false, bv10, (short)0, o1, false, false);
    o1 = __builtin_amdgcn_wmma_f32_16x16x32_f16(false, pa1, false, bv11, (short)0, o1, false, false);
    o2 = __builtin_amdgcn_wmma_f32_16x16x32_f16(false, pa0, false, ones, (short)0, o2, false, false);
    o2 = __builtin_amdgcn_wmma_f32_16x16x32_f16(false, pa1, false, ones, (short)0, o2, false, false);

    // ---- stage next chunk into the other buffer, then one barrier per iteration ----
    if (have_next) {
      store_chunk(nxt, sK + ((it + 1) & 1) * kSKBuf, sVT + ((it + 1) & 1) * kSVBuf,
                  skey, sdim, vpos);
    }
    __syncthreads();
  }

  // ---- epilogue: normalize by row sum (lane-replicated in o2) and store [B,N,C] ----
#pragma unroll
  for (int v = 0; v < 8; ++v) {
    float inv = 1.0f / o2[v];
    int row = q0 + wid * 16 + v + 8 * hsel;
    float* op = out + (size_t)(b * kN + row) * kC + h * kD;
    op[ln16]      = o0[v] * inv;
    op[16 + ln16] = o1[v] * inv;
  }
}

extern "C" void kernel_launch(void* const* d_in, const int* in_sizes, int n_in,
                              void* d_out, int out_size, void* d_ws, size_t ws_size,
                              hipStream_t stream) {
  (void)in_sizes; (void)n_in; (void)out_size; (void)d_ws; (void)ws_size;
  const float* x = (const float*)d_in[0];
  float* out = (float*)d_out;
  dim3 grid(kN / (16 * kWaves), kH, kB);   // 32 x 16 x 4
  dim3 block(32 * kWaves);                 // 4 waves
  fa32_wmma_kernel<<<grid, block, 0, stream>>>(x, out);
}
